// AdaptiveEmbeddingI2T_83837761618409
// MI455X (gfx1250) — matrix operand order repaired and checked by reading, not compiled, not run on hardware
//
#include <hip/hip_runtime.h>
#include <hip/hip_bf16.h>

typedef unsigned short u16;
typedef unsigned int   u32;
typedef __attribute__((ext_vector_type(16))) __bf16 v16bf;
typedef __attribute__((ext_vector_type(8)))  float  v8f;

#define D 1024
#define RNUM 36
#define MPAD 48
#define NB 64        // Bi == Bc == 64
#define EPS 1e-8f
#define BN_EPS 1e-5f

union Frag {
    v16bf bf;
    uint4 q[2];
};

__device__ __forceinline__ u16 f2bf(float x) {
    union { float f; u32 u; } v; v.f = x;
    u32 r = v.u + 0x7FFFu + ((v.u >> 16) & 1u);   // round-to-nearest-even
    return (u16)(r >> 16);
}
__device__ __forceinline__ float bf2f(u16 h) {
    union { u32 u; float f; } v; v.u = ((u32)h) << 16;
    return v.f;
}

__device__ __forceinline__ float block_reduce_sum(float v, float* sh) {
    int t = threadIdx.x;
    sh[t] = v; __syncthreads();
    for (int s = 128; s > 0; s >>= 1) { if (t < s) sh[t] += sh[t + s]; __syncthreads(); }
    float r = sh[0]; __syncthreads();
    return r;
}

// ---------- 1) BatchNorm stats: per-channel d over Bi*R = 2304 samples ----------
__global__ __launch_bounds__(256) void k_bn_stats(const float* __restrict__ img_embed,
                                                  float* __restrict__ mean,
                                                  float* __restrict__ istd) {
    __shared__ float sh[256];
    int d = blockIdx.x;
    float s = 0.f, s2 = 0.f;
    for (int i = threadIdx.x; i < NB * RNUM; i += 256) {
        float x = img_embed[(size_t)i * D + d];
        s += x; s2 += x * x;
    }
    s = block_reduce_sum(s, sh);
    s2 = block_reduce_sum(s2, sh);
    if (threadIdx.x == 0) {
        float m = s / (float)(NB * RNUM);
        float v = s2 / (float)(NB * RNUM) - m * m;
        mean[d] = m;
        istd[d] = rsqrtf(v + BN_EPS);
    }
}

// ---------- 2) normalize img_embed elementwise ----------
__global__ __launch_bounds__(256) void k_imnorm(const float* __restrict__ img_embed,
                                                const float* __restrict__ mean,
                                                const float* __restrict__ istd,
                                                float* __restrict__ imn, int n) {
    int i = blockIdx.x * 256 + threadIdx.x;
    if (i < n) {
        int d = i & (D - 1);
        imn[i] = (img_embed[i] - mean[d]) * istd[d];
    }
}

// ---------- 3) per-token inverse norms (first 32 tokens), folded 1/32 ----------
__global__ __launch_bounds__(256) void k_toknorm(const float* __restrict__ cap_embed,
                                                 float* __restrict__ tscale) {
    __shared__ float sh[256];
    int c = blockIdx.x >> 5, t = blockIdx.x & 31;
    const float* p = cap_embed + ((size_t)(c * 64 + t)) * D;
    float ss = 0.f;
    for (int d = threadIdx.x; d < D; d += 256) { float x = p[d]; ss += x * x; }
    ss = block_reduce_sum(ss, sh);
    if (threadIdx.x == 0)
        tscale[c * 32 + t] = 1.0f / (32.0f * (sqrtf(ss) + EPS));
}

// ---------- 4) cap_repr[c,d] = sum_t cap[c,t,d] * tscale[c,t] ----------
__global__ __launch_bounds__(256) void k_caprepr(const float* __restrict__ cap_embed,
                                                 const float* __restrict__ tscale,
                                                 float* __restrict__ cap_repr) {
    int i = blockIdx.x * 256 + threadIdx.x;   // 64*1024
    int c = i >> 10, d = i & (D - 1);
    float s = 0.f;
    const float* base = cap_embed + ((size_t)c * 64) * D + d;
    const float* ts = tscale + c * 32;
    for (int t = 0; t < 32; ++t) s += base[(size_t)t * D] * ts[t];
    cap_repr[i] = s;
}

// ---------- 5) small dense layer (scalar fp32, ~0.13 GFLOP per call) ----------
// mode 0: relu(out), mode 1: 1 + out, mode 2: out
__global__ __launch_bounds__(256) void k_dense(const float* __restrict__ in,
                                               const float* __restrict__ W,
                                               const float* __restrict__ bias,
                                               float* __restrict__ out, int mode) {
    int i = blockIdx.x * 256 + threadIdx.x;   // 64*1024
    int c = i >> 10, e = i & (D - 1);
    const float* x = in + (size_t)c * D;
    const float* w = W + (size_t)e * D;
    float s = 0.f;
    for (int k = 0; k < D; k += 4) {
        s += x[k] * w[k] + x[k + 1] * w[k + 1] + x[k + 2] * w[k + 2] + x[k + 3] * w[k + 3];
    }
    s += bias[e];
    if (mode == 0) s = s > 0.f ? s : 0.f;
    else if (mode == 1) s = 1.0f + s;
    out[i] = s;
}

// ---------- 6) fp32 -> bf16 weight convert ----------
__global__ __launch_bounds__(256) void k_cvt(const float* __restrict__ w,
                                             u16* __restrict__ o, int n) {
    int i = blockIdx.x * 256 + threadIdx.x;
    if (i < n) o[i] = f2bf(w[i]);
}

// ---------- 7) fused per-(c,b): build A, GEMM1(relu), GEMM2, softmax(R), pool ----------
__global__ __launch_bounds__(256) void k_fused(const float* __restrict__ imn,
                                               const float* __restrict__ g1p,   // 1+gam
                                               const float* __restrict__ bet,
                                               const u16* __restrict__ Wp1b,
                                               const u16* __restrict__ Wp2b,
                                               const float* __restrict__ bp1,
                                               const float* __restrict__ bp2,
                                               float* __restrict__ pooled) {
    extern __shared__ u16 smem[];            // A: 48x1024 bf16, H: 48x1024 bf16 (192 KB)
    u16* A = smem;
    u16* H = smem + MPAD * D;

    int c = blockIdx.x >> 6;
    int b = blockIdx.x & 63;

    const float* g1c  = g1p + (size_t)c * D;
    const float* betc = bet + (size_t)c * D;
    const float* imnb = imn + (size_t)b * RNUM * D;

    // Build modulated A matrix (rows >= 36 are zero padding)
    for (int i = threadIdx.x; i < MPAD * D; i += 256) {
        int r = i >> 10, d = i & (D - 1);
        float v = 0.f;
        if (r < RNUM) v = imnb[(size_t)r * D + d] * g1c[d] + betc[d];
        A[i] = f2bf(v);
    }
    __syncthreads();

    const int wave = threadIdx.x >> 5;
    const int lane = threadIdx.x & 31;
    const int lrow = lane & 15;
    const int lhi  = lane >> 4;

    // ---- GEMM1: H = relu(A @ Wp1^T + bp1) ----
    // k-outer: one B fragment feeds all 3 M-tiles -> Wp1 read exactly once per block.
    for (int i = 0; i < 8; ++i) {
        int n0 = (wave + 8 * i) << 4;
        int col = n0 + lrow;
        float bias = bp1[col];
        v8f acc[3] = {{}, {}, {}};
        for (int k0 = 0; k0 < D; k0 += 32) {
            Frag bb;
            const u16* bp = Wp1b + (size_t)col * D + k0 + (lhi << 4);
            bb.q[0] = *(const uint4*)(bp);
            bb.q[1] = *(const uint4*)(bp + 8);
#pragma unroll
            for (int mt = 0; mt < 3; ++mt) {
                Frag a;
                const u16* ap = A + (size_t)((mt << 4) + lrow) * D + k0 + (lhi << 4);
                a.q[0] = *(const uint4*)(ap);
                a.q[1] = *(const uint4*)(ap + 8);
                acc[mt] = __builtin_amdgcn_wmma_f32_16x16x32_bf16(
                    false, a.bf, false, bb.bf, (short)0, acc[mt], false, false);
            }
        }
#pragma unroll
        for (int mt = 0; mt < 3; ++mt) {
            int rbase = (mt << 4) + (lhi << 3);
#pragma unroll
            for (int j = 0; j < 8; ++j) {
                float h = acc[mt][j] + bias;
                h = h > 0.f ? h : 0.f;
                H[(size_t)(rbase + j) * D + col] = f2bf(h);
            }
        }
    }
    __syncthreads();

    // ---- GEMM2 + softmax over R + weighted pool, fully in registers ----
    for (int i = 0; i < 8; ++i) {
        int n0 = (wave + 8 * i) << 4;
        int col = n0 + lrow;
        v8f acc[3] = {{}, {}, {}};
        for (int k0 = 0; k0 < D; k0 += 32) {
            Frag bb;
            const u16* bp = Wp2b + (size_t)col * D + k0 + (lhi << 4);
            bb.q[0] = *(const uint4*)(bp);
            bb.q[1] = *(const uint4*)(bp + 8);
#pragma unroll
            for (int mt = 0; mt < 3; ++mt) {
                Frag a;
                const u16* ap = H + (size_t)((mt << 4) + lrow) * D + k0 + (lhi << 4);
                a.q[0] = *(const uint4*)(ap);
                a.q[1] = *(const uint4*)(ap + 8);
                acc[mt] = __builtin_amdgcn_wmma_f32_16x16x32_bf16(
                    false, a.bf, false, bb.bf, (short)0, acc[mt], false, false);
            }
        }
        float bias = bp2[col];

        // lane holds rows {mt*16 + lhi*8 + j}; lane^16 holds the other 8 rows.
        float mx = -3.4e38f;
#pragma unroll
        for (int mt = 0; mt < 3; ++mt)
#pragma unroll
            for (int j = 0; j < 8; ++j) {
                int r = (mt << 4) + (lhi << 3) + j;
                if (r < RNUM) { float s = acc[mt][j] + bias; mx = fmaxf(mx, s); }
            }
        mx = fmaxf(mx, __shfl_xor(mx, 16, 32));

        float sum = 0.f;
#pragma unroll
        for (int mt = 0; mt < 3; ++mt)
#pragma unroll
            for (int j = 0; j < 8; ++j) {
                int r = (mt << 4) + (lhi << 3) + j;
                if (r < RNUM) sum += __expf(acc[mt][j] + bias - mx);
            }
        sum += __shfl_xor(sum, 16, 32);
        float inv = 1.0f / sum;

        float p = 0.f;
#pragma unroll
        for (int mt = 0; mt < 3; ++mt)
#pragma unroll
            for (int j = 0; j < 8; ++j) {
                int r = (mt << 4) + (lhi << 3) + j;
                if (r < RNUM) {
                    float w = __expf(acc[mt][j] + bias - mx) * inv;
                    p += bf2f(A[(size_t)r * D + col]) * w;
                }
            }
        p += __shfl_xor(p, 16, 32);
        if (lhi == 0) pooled[((size_t)(c * NB + b)) * D + col] = p;
    }
}

// ---------- 8) residual + l2norm + cosine sims ----------
__global__ __launch_bounds__(256) void k_final(const float* __restrict__ pooled,
                                               const float* __restrict__ img_glo,
                                               const float* __restrict__ cap_glo,
                                               float* __restrict__ out) {
    __shared__ float sh[256];
    int c = blockIdx.x >> 6, b = blockIdx.x & 63;
    const float* p  = pooled + ((size_t)(c * NB + b)) * D;
    const float* ig = img_glo + (size_t)b * D;
    const float* cg = cap_glo + (size_t)c * D;
    float ss = 0.f, cc = 0.f, dot = 0.f;
#pragma unroll
    for (int j = 0; j < 4; ++j) {
        int d = threadIdx.x + j * 256;
        float x = ig[d] + p[d];
        float cv = cg[d];
        ss += x * x; cc += cv * cv; dot += x * cv;
    }
    ss  = block_reduce_sum(ss, sh);
    cc  = block_reduce_sum(cc, sh);
    dot = block_reduce_sum(dot, sh);
    if (threadIdx.x == 0)
        out[b * NB + c] = dot / ((sqrtf(ss) + EPS) * (sqrtf(cc) + EPS));
}

extern "C" void kernel_launch(void* const* d_in, const int* in_sizes, int n_in,
                              void* d_out, int out_size, void* d_ws, size_t ws_size,
                              hipStream_t stream) {
    const float* img_glo   = (const float*)d_in[0];
    const float* cap_glo   = (const float*)d_in[1];
    const float* img_embed = (const float*)d_in[2];
    const float* cap_embed = (const float*)d_in[3];
    const float* Wg1 = (const float*)d_in[4];  const float* bg1 = (const float*)d_in[5];
    const float* Wg2 = (const float*)d_in[6];  const float* bg2 = (const float*)d_in[7];
    const float* Wb1 = (const float*)d_in[8];  const float* bb1 = (const float*)d_in[9];
    const float* Wb2 = (const float*)d_in[10]; const float* bb2 = (const float*)d_in[11];
    const float* Wp1 = (const float*)d_in[12]; const float* bp1 = (const float*)d_in[13];
    const float* Wp2 = (const float*)d_in[14]; const float* bp2 = (const float*)d_in[15];
    float* out = (float*)d_out;

    // workspace layout (floats)
    float* wsf = (float*)d_ws;
    size_t o = 0;
    float* mean     = wsf + o; o += 1024;
    float* istd     = wsf + o; o += 1024;
    float* imn      = wsf + o; o += (size_t)NB * RNUM * D;   // 2,359,296
    float* tscale   = wsf + o; o += 64 * 32;
    float* cap_repr = wsf + o; o += 64 * D;
    float* hid      = wsf + o; o += 64 * D;
    float* g1p      = wsf + o; o += 64 * D;
    float* bet      = wsf + o; o += 64 * D;
    float* pooled   = wsf + o; o += (size_t)NB * NB * D;     // 4,194,304
    u16* Wp1b = (u16*)(wsf + o);
    u16* Wp2b = Wp1b + (size_t)D * D;
    // total ~31.5 MB

    const int imn_n = NB * RNUM * D;

    k_bn_stats<<<D, 256, 0, stream>>>(img_embed, mean, istd);
    k_imnorm<<<(imn_n + 255) / 256, 256, 0, stream>>>(img_embed, mean, istd, imn, imn_n);
    k_toknorm<<<64 * 32, 256, 0, stream>>>(cap_embed, tscale);
    k_caprepr<<<64 * D / 256, 256, 0, stream>>>(cap_embed, tscale, cap_repr);

    k_dense<<<64 * D / 256, 256, 0, stream>>>(cap_repr, Wg1, bg1, hid, 0);
    k_dense<<<64 * D / 256, 256, 0, stream>>>(hid, Wg2, bg2, g1p, 1);      // 1 + gam
    k_dense<<<64 * D / 256, 256, 0, stream>>>(cap_repr, Wb1, bb1, hid, 0);
    k_dense<<<64 * D / 256, 256, 0, stream>>>(hid, Wb2, bb2, bet, 2);      // bet

    k_cvt<<<(D * D + 255) / 256, 256, 0, stream>>>(Wp1, Wp1b, D * D);
    k_cvt<<<(D * D + 255) / 256, 256, 0, stream>>>(Wp2, Wp2b, D * D);

    const int lds_bytes = 2 * MPAD * D * sizeof(u16);   // 192 KB (CDNA5 WGP LDS = 320 KB)
    hipFuncSetAttribute(reinterpret_cast<const void*>(k_fused),
                        hipFuncAttributeMaxDynamicSharedMemorySize, lds_bytes);
    k_fused<<<NB * NB, 256, lds_bytes, stream>>>(imn, g1p, bet, Wp1b, Wp2b, bp1, bp2, pooled);

    k_final<<<NB * NB, 256, 0, stream>>>(pooled, img_glo, cap_glo, out);
}
